// GraphTGI_hetero_34385508171920
// MI455X (gfx1250) — compile-verified, matrix operand order checked by reading, not run on hardware
//
#include <hip/hip_runtime.h>

// ---------------------------------------------------------------------------
// MI455X (gfx1250) heterogeneous-GNN forward pass.
//  * all dense matmuls: v_wmma_f32_16x16x32_bf16, f32 accumulate
//  * weights pre-transposed to [N][K] bf16 so both GEMM operands stage as
//    contiguous rows -> GLOBAL_LOAD_ASYNC_TO_LDS_B128 (ASYNCcnt), LDS
//    double-buffered to overlap DMA with the XDL pipe
//  * B-fragment loads software-pipelined (rotation depth 2) so each WMMA's
//    LDS dependency is issued two steps ahead -> no per-WMMA dscnt-0 stall
//  * degree-norm row scalars commute with the matmul -> folded into epilogues
//  * edge aggregation: global f32 atomics; activations fused into combines
// ---------------------------------------------------------------------------

#define SLOPE 0.2f

typedef __bf16 bf16;
typedef __attribute__((ext_vector_type(16))) __bf16 v16bf;
typedef __attribute__((ext_vector_type(8)))  __bf16 v8bf;
typedef __attribute__((ext_vector_type(8)))  float  v8f;

__device__ __forceinline__ float lrelu(float v) { return v > 0.0f ? v : v * SLOPE; }

// ---------------------------- elementwise ----------------------------------

// f32 [rowsS, colsS] -> bf16 [rowsD, colsD] row-major, zero pad.
__global__ void cvt_pad_kernel(const float* __restrict__ src, bf16* __restrict__ dst,
                               int rowsS, int colsS, int colsD, long total) {
  long i = (long)blockIdx.x * blockDim.x + threadIdx.x;
  if (i >= total) return;
  int r = (int)(i / colsD), c = (int)(i % colsD);
  float v = (r < rowsS && c < colsS) ? src[(long)r * colsS + c] : 0.0f;
  dst[i] = (bf16)v;
}

// f32 [K, N] -> bf16 [N, Kp] (transposed, K zero-padded to Kp).
__global__ void cvt_pad_t_kernel(const float* __restrict__ src, bf16* __restrict__ dst,
                                 int K, int N, int Kp, long total) {
  long i = (long)blockIdx.x * blockDim.x + threadIdx.x;
  if (i >= total) return;
  int n = (int)(i / Kp), k = (int)(i % Kp);
  float v = (k < K) ? src[(long)k * N + n] : 0.0f;
  dst[i] = (bf16)v;
}

__global__ void deg_count_kernel(const int* __restrict__ idx, float* __restrict__ deg, int n) {
  int i = blockIdx.x * blockDim.x + threadIdx.x;
  if (i < n) atomicAdd(&deg[idx[i]], 1.0f);
}

__global__ void rsqrt_clip_kernel(float* __restrict__ d, int n) {
  int i = blockIdx.x * blockDim.x + threadIdx.x;
  if (i < n) d[i] = rsqrtf(fmaxf(d[i], 1.0f));
}

// agg[dst] += feat[src] over edges; one block per edge, float4 chunks.
__global__ void scatter_add_kernel(const float* __restrict__ feat,
                                   const int* __restrict__ src, const int* __restrict__ dst,
                                   int nE, int S4, float* __restrict__ agg) {
  int e = blockIdx.x;
  if (e >= nE) return;
  long s = src[e], d = dst[e];
  const float4* fs = (const float4*)feat + s * S4;
  float* ad = agg + d * (long)S4 * 4;
  for (int f = threadIdx.x; f < S4; f += blockDim.x) {
    float4 v = fs[f];
    atomicAdd(&ad[f * 4 + 0], v.x);
    atomicAdd(&ad[f * 4 + 1], v.y);
    atomicAdd(&ad[f * 4 + 2], v.z);
    atomicAdd(&ad[f * 4 + 3], v.w);
  }
}

// v = lrelu(agg*iscale + bias) [+ add] [; extra lrelu] -> f32 and/or bf16
__global__ void combine1_kernel(const float* __restrict__ agg, const float* __restrict__ iscale,
                                const float* __restrict__ bias, const float* __restrict__ add,
                                int extraAct, float* __restrict__ outF, bf16* __restrict__ outB,
                                int n, int S) {
  long i = (long)blockIdx.x * blockDim.x + threadIdx.x;
  if (i >= (long)n * S) return;
  int r = (int)(i / S), c = (int)(i % S);
  float v = agg[i];
  if (iscale) v *= iscale[r];
  if (bias)   v += bias[c];
  v = lrelu(v);
  if (add) v += add[i];
  if (extraAct) v = lrelu(v);
  if (outF) outF[i] = v;
  if (outB) outB[i] = (bf16)v;
}

// v = lrelu(aggA*isA+bA) + lrelu(aggB*isB+bB) [; extra lrelu]
__global__ void combine2_kernel(const float* __restrict__ aggA, const float* __restrict__ isA,
                                const float* __restrict__ bA,
                                const float* __restrict__ aggB, const float* __restrict__ isB,
                                const float* __restrict__ bB,
                                int extraAct, float* __restrict__ outF, bf16* __restrict__ outB,
                                int n, int S) {
  long i = (long)blockIdx.x * blockDim.x + threadIdx.x;
  if (i >= (long)n * S) return;
  int r = (int)(i / S), c = (int)(i % S);
  float v = lrelu(aggA[i] * isA[r] + bA[c]) + lrelu(aggB[i] * isB[r] + bB[c]);
  if (extraAct) v = lrelu(v);
  if (outF) outF[i] = v;
  if (outB) outB[i] = (bf16)v;
}

// ------------------------------ WMMA GEMM -----------------------------------
// C[M,N] = epilogue( A[M,K] @ Bt[N,K]^T ) ; A,Bt bf16 row-major, K%64==0,
// N%128==0. 256 threads = 8 waves; tile 128(M) x 128(N) x 64(K). Each wave
// owns a 16x128 strip -> 16 v_wmma per K-step. Staging is async global->LDS
// (b128 per lane), double buffered: DMA of tile k+1 overlaps WMMA on tile k.

#define TM 128
#define TN 128
#define TK 64

__device__ __forceinline__ void async_b128(unsigned lds_off, const bf16* g) {
  asm volatile("global_load_async_to_lds_b128 %0, %1, off"
               :: "v"(lds_off), "v"(g) : "memory");
}
__device__ __forceinline__ void wait_async0() {
  asm volatile("s_wait_asynccnt 0x0" ::: "memory");
}

__global__ __launch_bounds__(256) void gemm_bf16_wmma_kernel(
    const bf16* __restrict__ A, const bf16* __restrict__ Bt,
    int M, int N, int K,
    const float* __restrict__ rowScale, const float* __restrict__ bias,
    int act, float* __restrict__ outF, bf16* __restrict__ outB) {
  __shared__ alignas(16) bf16 sA[2][TM][TK];  // 2 x 16 KB
  __shared__ alignas(16) bf16 sB[2][TN][TK];  // 2 x 16 KB

  const int tid  = threadIdx.x;
  const int wave = tid >> 5;
  const int lane = tid & 31;
  const int la   = lane & 15;
  const int khalf = (lane >> 4) << 3;  // 0 or 8 per bf16 16x32 fragment layout
  const int m0 = blockIdx.x * TM;
  const int n0 = blockIdx.y * TN;

  const v8f vz = {0.f, 0.f, 0.f, 0.f, 0.f, 0.f, 0.f, 0.f};
  v8f acc[8];
#pragma unroll
  for (int t = 0; t < 8; ++t) acc[t] = vz;

  // staging map: thread covers (row r, 32-elem half h) of both A and B tiles.
  const int r = tid >> 1;
  const int h = (tid & 1) << 5;
  const int garow = (m0 + r < M) ? (m0 + r) : (M - 1);  // clamp: rows >= M are
  const bf16* gA = A  + (long)garow * K + h;            // computed then masked
  const bf16* gB = Bt + (long)(n0 + r) * K + h;

  auto stage = [&](int buf, int k0) {
    unsigned lA = (unsigned)(size_t)&sA[buf][r][h];
    unsigned lB = (unsigned)(size_t)&sB[buf][r][h];
    const bf16* pa = gA + k0;
    const bf16* pb = gB + k0;
#pragma unroll
    for (int c = 0; c < 4; ++c) {  // 4 x 16B = 32 elems per tile per thread
      async_b128(lA + c * 16, pa + c * 8);
      async_b128(lB + c * 16, pb + c * 8);
    }
  };

  const int mrow = (wave << 4) + la;

  // load B fragment for column-tile t at K offset kk
  auto loadB = [&](int buf, int kk, int t) -> v16bf {
    int col = (t << 4) + la;
    v8bf b0 = *(const v8bf*)&sB[buf][col][kk + khalf];
    v8bf b1 = *(const v8bf*)&sB[buf][col][kk + 16 + khalf];
    return __builtin_shufflevector(b0, b1, 0, 1, 2, 3, 4, 5, 6, 7,
                                   8, 9, 10, 11, 12, 13, 14, 15);
  };

  auto compute = [&](int buf) {
#pragma unroll
    for (int kk = 0; kk < TK; kk += 32) {
      // A fragment for this wave's 16-row strip
      v8bf a0 = *(const v8bf*)&sA[buf][mrow][kk + khalf];
      v8bf a1 = *(const v8bf*)&sA[buf][mrow][kk + 16 + khalf];
      v16bf af = __builtin_shufflevector(a0, a1, 0, 1, 2, 3, 4, 5, 6, 7,
                                         8, 9, 10, 11, 12, 13, 14, 15);
      // software-pipelined B fragments: issue load t+2 before WMMA t so each
      // WMMA's LDS dependency is already 2 steps old (no dscnt-0 stall).
      v16bf bq0 = loadB(buf, kk, 0);
      v16bf bq1 = loadB(buf, kk, 1);
#pragma unroll
      for (int t = 0; t < 8; ++t) {
        v16bf bnew = (t + 2 < 8) ? loadB(buf, kk, t + 2) : bq0;
        acc[t] = __builtin_amdgcn_wmma_f32_16x16x32_bf16(
            false, af, false, bq0, (short)0, acc[t], false, false);
        bq0 = bq1;
        bq1 = bnew;
      }
    }
  };

  stage(0, 0);
  wait_async0();
  __syncthreads();
  int buf = 0;
  for (int k0 = 0; k0 < K; k0 += TK) {
    if (k0 + TK < K) stage(buf ^ 1, k0 + TK);  // DMA next tile
    compute(buf);                              // 16 WMMAs on current tile
    wait_async0();
    __syncthreads();
    buf ^= 1;
  }

  // C/D layout: VGPR j -> M = j (+8 for lanes 16-31); N = lane & 15
  const int mbase = m0 + (wave << 4) + ((lane >> 4) << 3);
#pragma unroll
  for (int t = 0; t < 8; ++t) {
#pragma unroll
    for (int j = 0; j < 8; ++j) {
      int rr = mbase + j;
      int cc = n0 + (t << 4) + la;
      if (rr < M) {
        float v = acc[t][j];
        if (rowScale) v *= rowScale[rr];
        if (bias)     v += bias[cc];
        if (act)      v = lrelu(v);
        long o = (long)rr * N + cc;
        if (outF) outF[o] = v;
        if (outB) outB[o] = (bf16)v;
      }
    }
  }
}

// ------------------------------ orchestration -------------------------------

extern "C" void kernel_launch(void* const* d_in, const int* in_sizes, int n_in,
                              void* d_out, int out_size, void* d_ws, size_t ws_size,
                              hipStream_t stream) {
  constexpr int N_TF = 20000, N_TG = 50000, N_DIS = 20000, N_GO = 50000;
  constexpr int IN = 512, GO_IN = 4338, GO_INP = 4352, S1 = 1024, S2 = 512, S3 = 256;
  (void)n_in; (void)out_size; (void)ws_size;

  const float* x_TF = (const float*)d_in[0];
  const float* x_tg = (const float*)d_in[1];
  const float* x_dis = (const float*)d_in[2];
  const float* x_go = (const float*)d_in[3];
  const float* wl_TF = (const float*)d_in[4];  const float* bl_TF = (const float*)d_in[5];
  const float* wl_tg = (const float*)d_in[6];  const float* bl_tg = (const float*)d_in[7];
  const float* wl_dis = (const float*)d_in[8]; const float* bl_dis = (const float*)d_in[9];
  const float* wl_go = (const float*)d_in[10]; const float* bl_go = (const float*)d_in[11];
  const float* w1 = (const float*)d_in[12];    const float* b1 = (const float*)d_in[13];
  const float* w2 = (const float*)d_in[14];    const float* b2 = (const float*)d_in[15];
  const int* reg_src = (const int*)d_in[16];   const int* reg_dst = (const int*)d_in[17];
  const int* a1_src = (const int*)d_in[18];    const int* a1_dst = (const int*)d_in[19];
  const int* a2_src = (const int*)d_in[20];    const int* a2_dst = (const int*)d_in[21];
  const int* a3_src = (const int*)d_in[22];    const int* a3_dst = (const int*)d_in[23];
  const int nREG = in_sizes[16], nA1 = in_sizes[18], nA2 = in_sizes[20], nA3 = in_sizes[22];

  // ---- workspace bump allocator ----
  char* cur = (char*)d_ws;
  auto alloc = [&](size_t bytes) -> void* {
    void* p = cur; cur += (bytes + 255) & ~(size_t)255; return p;
  };
  // bf16 GEMM-A inputs (x_go K-padded 4338 -> 4352)
  bf16* bxTF  = (bf16*)alloc((size_t)N_TF * IN * 2);
  bf16* bxtg  = (bf16*)alloc((size_t)N_TG * IN * 2);
  bf16* bxdis = (bf16*)alloc((size_t)N_DIS * IN * 2);
  bf16* bxgo  = (bf16*)alloc((size_t)N_GO * GO_INP * 2);
  // bf16 weights, TRANSPOSED to [N][Kp]
  bf16* bwlTFt  = (bf16*)alloc((size_t)S1 * IN * 2);
  bf16* bwltgt  = (bf16*)alloc((size_t)S1 * IN * 2);
  bf16* bwldist = (bf16*)alloc((size_t)S1 * IN * 2);
  bf16* bwlgot  = (bf16*)alloc((size_t)S1 * GO_INP * 2);
  bf16* bw1t = (bf16*)alloc((size_t)6 * S2 * S1 * 2);
  bf16* bw2t = (bf16*)alloc((size_t)6 * S3 * S2 * 2);
  // bf16 layer activations
  bf16* bhTF1  = (bf16*)alloc((size_t)N_TF * S1 * 2);
  bf16* bhtg1  = (bf16*)alloc((size_t)N_TG * S1 * 2);
  bf16* bhdis1 = (bf16*)alloc((size_t)N_DIS * S1 * 2);
  bf16* bhgo1  = (bf16*)alloc((size_t)N_GO * S1 * 2);
  bf16* bhTF2  = (bf16*)alloc((size_t)N_TF * S2 * 2);
  bf16* bhtg2  = (bf16*)alloc((size_t)N_TG * S2 * 2);
  bf16* bhdis2 = (bf16*)alloc((size_t)N_DIS * S2 * 2);
  bf16* bhgo2  = (bf16*)alloc((size_t)N_GO * S2 * 2);
  // degree rsqrt buffers (contiguous: 280000 floats)
  float* degs = (float*)alloc((size_t)280000 * 4);
  float* odeg_reg = degs;
  float* ideg_reg = odeg_reg + N_TF;
  float* odeg_a1  = ideg_reg + N_TG;
  float* ideg_a1  = odeg_a1 + N_TG;
  float* odeg_a2  = ideg_a1 + N_DIS;
  float* ideg_a2  = odeg_a2 + N_DIS;
  float* odeg_a3  = ideg_a2 + N_TG;
  float* ideg_a3  = odeg_a3 + N_GO;
  // f32 scratch (worst case 50000 x 512)
  float* feat    = (float*)alloc((size_t)N_GO * S2 * 4);
  float* aggA    = (float*)alloc((size_t)N_TG * S2 * 4);
  float* aggB    = (float*)alloc((size_t)N_TG * S2 * 4);
  float* selfbuf = (float*)alloc((size_t)N_GO * S2 * 4);

  // ---- launch helpers ----
  auto cvt = [&](const float* s, bf16* d, int rowsS, int colsS, int rowsD, int colsD) {
    long total = (long)rowsD * colsD;
    cvt_pad_kernel<<<(int)((total + 255) / 256), 256, 0, stream>>>(s, d, rowsS, colsS, colsD, total);
  };
  auto cvtT = [&](const float* s, bf16* d, int K, int N, int Kp) {
    long total = (long)N * Kp;
    cvt_pad_t_kernel<<<(int)((total + 255) / 256), 256, 0, stream>>>(s, d, K, N, Kp, total);
  };
  auto gemm = [&](const bf16* A, const bf16* Bt, int M, int N, int K,
                  const float* rs, const float* bias, int act, float* oF, bf16* oB) {
    dim3 grid((M + TM - 1) / TM, N / TN);
    gemm_bf16_wmma_kernel<<<grid, 256, 0, stream>>>(A, Bt, M, N, K, rs, bias, act, oF, oB);
  };
  auto zero = [&](void* p, size_t bytes) { hipMemsetAsync(p, 0, bytes, stream); };
  auto scatter = [&](const float* f, const int* s, const int* d, int nE, int S, float* agg) {
    scatter_add_kernel<<<nE, 128, 0, stream>>>(f, s, d, nE, S / 4, agg);
  };
  auto comb1 = [&](const float* agg, const float* is, const float* b, const float* add,
                   int ea, float* oF, bf16* oB, int n, int S) {
    long total = (long)n * S;
    combine1_kernel<<<(int)((total + 255) / 256), 256, 0, stream>>>(agg, is, b, add, ea, oF, oB, n, S);
  };
  auto comb2 = [&](const float* aA, const float* iA, const float* bA,
                   const float* aB, const float* iB, const float* bB,
                   int ea, float* oF, bf16* oB, int n, int S) {
    long total = (long)n * S;
    combine2_kernel<<<(int)((total + 255) / 256), 256, 0, stream>>>(aA, iA, bA, aB, iB, bB, ea, oF, oB, n, S);
  };
  auto degcnt = [&](const int* idx, float* deg, int n) {
    deg_count_kernel<<<(n + 255) / 256, 256, 0, stream>>>(idx, deg, n);
  };

  // ---- phase 0: degrees + bf16 conversion ----
  zero(degs, (size_t)280000 * 4);
  degcnt(reg_src, odeg_reg, nREG); degcnt(reg_dst, ideg_reg, nREG);
  degcnt(a1_src, odeg_a1, nA1);    degcnt(a1_dst, ideg_a1, nA1);
  degcnt(a2_src, odeg_a2, nA2);    degcnt(a2_dst, ideg_a2, nA2);
  degcnt(a3_src, odeg_a3, nA3);    degcnt(a3_dst, ideg_a3, nA3);
  rsqrt_clip_kernel<<<(280000 + 255) / 256, 256, 0, stream>>>(degs, 280000);

  cvt(x_TF, bxTF, N_TF, IN, N_TF, IN);
  cvt(x_tg, bxtg, N_TG, IN, N_TG, IN);
  cvt(x_dis, bxdis, N_DIS, IN, N_DIS, IN);
  cvt(x_go, bxgo, N_GO, GO_IN, N_GO, GO_INP);   // column pad
  cvtT(wl_TF, bwlTFt, IN, S1, IN);
  cvtT(wl_tg, bwltgt, IN, S1, IN);
  cvtT(wl_dis, bwldist, IN, S1, IN);
  cvtT(wl_go, bwlgot, GO_IN, S1, GO_INP);       // K pad in transposed layout
  for (int rsl = 0; rsl < 6; ++rsl) {
    cvtT(w1 + (size_t)rsl * S1 * S2, bw1t + (size_t)rsl * S2 * S1, S1, S2, S1);
    cvtT(w2 + (size_t)rsl * S2 * S3, bw2t + (size_t)rsl * S3 * S2, S2, S3, S2);
  }

  // ---- phase 1: HeteroLinear -> S1 (bf16 activations) ----
  gemm(bxTF, bwlTFt, N_TF, S1, IN, nullptr, bl_TF, 0, nullptr, bhTF1);
  gemm(bxtg, bwltgt, N_TG, S1, IN, nullptr, bl_tg, 0, nullptr, bhtg1);
  gemm(bxdis, bwldist, N_DIS, S1, IN, nullptr, bl_dis, 0, nullptr, bhdis1);
  gemm(bxgo, bwlgot, N_GO, S1, GO_INP, nullptr, bl_go, 0, nullptr, bhgo1);

  // ---- phase 2: conv1 (K=S1, out=S2), extra activation, bf16 out ----
  const size_t w1s = (size_t)S2 * S1;  // transposed slice stride
  // tg = gconv(hTF, w1[0], reg) + gconv(hdis, w1[2], a2)
  gemm(bhTF1, bw1t + 0 * w1s, N_TF, S2, S1, odeg_reg, nullptr, 0, feat, nullptr);
  zero(aggA, (size_t)N_TG * S2 * 4);
  scatter(feat, reg_src, reg_dst, nREG, S2, aggA);
  gemm(bhdis1, bw1t + 2 * w1s, N_DIS, S2, S1, odeg_a2, nullptr, 0, feat, nullptr);
  zero(aggB, (size_t)N_TG * S2 * 4);
  scatter(feat, a2_src, a2_dst, nA2, S2, aggB);
  comb2(aggA, ideg_reg, b1 + 0 * S2, aggB, ideg_a2, b1 + 2 * S2, 1, nullptr, bhtg2, N_TG, S2);
  // dis = gconv(htg, w1[1], a1)
  gemm(bhtg1, bw1t + 1 * w1s, N_TG, S2, S1, odeg_a1, nullptr, 0, feat, nullptr);
  zero(aggA, (size_t)N_DIS * S2 * 4);
  scatter(feat, a1_src, a1_dst, nA1, S2, aggA);
  comb1(aggA, ideg_a1, b1 + 1 * S2, nullptr, 1, nullptr, bhdis2, N_DIS, S2);
  // TF = gconv(hgo, w1[3], a3) + self_conv(hTF, w1[5])
  gemm(bhTF1, bw1t + 5 * w1s, N_TF, S2, S1, nullptr, b1 + 5 * S2, 1, selfbuf, nullptr);
  gemm(bhgo1, bw1t + 3 * w1s, N_GO, S2, S1, odeg_a3, nullptr, 0, feat, nullptr);
  zero(aggB, (size_t)N_TF * S2 * 4);
  scatter(feat, a3_src, a3_dst, nA3, S2, aggB);
  comb1(aggB, ideg_a3, b1 + 3 * S2, selfbuf, 1, nullptr, bhTF2, N_TF, S2);
  // go = self_conv(hgo, w1[4]); extra activation via comb1's inner lrelu
  gemm(bhgo1, bw1t + 4 * w1s, N_GO, S2, S1, nullptr, b1 + 4 * S2, 1, selfbuf, nullptr);
  comb1(selfbuf, nullptr, nullptr, nullptr, 0, nullptr, bhgo2, N_GO, S2);

  // ---- phase 3: conv2 (K=S2, out=S3) -> d_out f32, no extra activation ----
  float* outTF  = (float*)d_out;
  float* outtg  = outTF + (size_t)N_TF * S3;
  float* outdis = outtg + (size_t)N_TG * S3;
  float* outgo  = outdis + (size_t)N_DIS * S3;
  const size_t w2s = (size_t)S3 * S2;
  // tg
  gemm(bhTF2, bw2t + 0 * w2s, N_TF, S3, S2, odeg_reg, nullptr, 0, feat, nullptr);
  zero(aggA, (size_t)N_TG * S3 * 4);
  scatter(feat, reg_src, reg_dst, nREG, S3, aggA);
  gemm(bhdis2, bw2t + 2 * w2s, N_DIS, S3, S2, odeg_a2, nullptr, 0, feat, nullptr);
  zero(aggB, (size_t)N_TG * S3 * 4);
  scatter(feat, a2_src, a2_dst, nA2, S3, aggB);
  comb2(aggA, ideg_reg, b2 + 0 * S3, aggB, ideg_a2, b2 + 2 * S3, 0, outtg, nullptr, N_TG, S3);
  // dis
  gemm(bhtg2, bw2t + 1 * w2s, N_TG, S3, S2, odeg_a1, nullptr, 0, feat, nullptr);
  zero(aggA, (size_t)N_DIS * S3 * 4);
  scatter(feat, a1_src, a1_dst, nA1, S3, aggA);
  comb1(aggA, ideg_a1, b2 + 1 * S3, nullptr, 0, outdis, nullptr, N_DIS, S3);
  // TF
  gemm(bhTF2, bw2t + 5 * w2s, N_TF, S3, S2, nullptr, b2 + 5 * S3, 1, selfbuf, nullptr);
  gemm(bhgo2, bw2t + 3 * w2s, N_GO, S3, S2, odeg_a3, nullptr, 0, feat, nullptr);
  zero(aggB, (size_t)N_TF * S3 * 4);
  scatter(feat, a3_src, a3_dst, nA3, S3, aggB);
  comb1(aggB, ideg_a3, b2 + 3 * S3, selfbuf, 0, outTF, nullptr, N_TF, S3);
  // go -> straight into d_out
  gemm(bhgo2, bw2t + 4 * w2s, N_GO, S3, S2, nullptr, b2 + 4 * S3, 1, outgo, nullptr);
}
